// _ModConv_66932770340987
// MI455X (gfx1250) — matrix operand order repaired and checked by reading, not compiled, not run on hardware
//
#include <hip/hip_runtime.h>
#include <hip/hip_bf16.h>
#include <math.h>

typedef __attribute__((ext_vector_type(16))) _Float16 v16h;
typedef __attribute__((ext_vector_type(8)))  _Float16 v8h;
typedef __attribute__((ext_vector_type(8)))  float    v8f;

#define B_  8
#define I_  256
#define O_  256
#define K_  3
#define L_  8192
#define T_  128

// LDS row stride in halfs (80 bytes: 16B-aligned, spreads banks)
#define ASTRIDE 40
#define BSTRIDE 40

// ---------------------------------------------------------------------------
// Kernel 1: MLP  s[b,i] = (silu(silu(t@w1+b1)@w2+b2)@w3 + b3 + 1)
// one block per batch row
// ---------------------------------------------------------------------------
__device__ __forceinline__ float silu_f(float x) {
    return x / (1.0f + __expf(-x));
}

__global__ void __launch_bounds__(256)
modconv_mlp_kernel(const float* __restrict__ t,
                   const float* __restrict__ w1, const float* __restrict__ b1,
                   const float* __restrict__ w2, const float* __restrict__ b2,
                   const float* __restrict__ w3, const float* __restrict__ b3,
                   float* __restrict__ s_out) {
    __shared__ float bufA[T_];
    __shared__ float bufB[T_];
    const int b   = blockIdx.x;
    const int tid = threadIdx.x;

    if (tid < T_) bufA[tid] = t[b * T_ + tid];
    __syncthreads();

    if (tid < T_) {
        float acc = b1[tid];
        #pragma unroll 8
        for (int ii = 0; ii < T_; ++ii) acc += bufA[ii] * w1[ii * T_ + tid];
        bufB[tid] = silu_f(acc);
    }
    __syncthreads();

    if (tid < T_) {
        float acc = b2[tid];
        #pragma unroll 8
        for (int ii = 0; ii < T_; ++ii) acc += bufB[ii] * w2[ii * T_ + tid];
        bufA[tid] = silu_f(acc);
    }
    __syncthreads();

    if (tid < I_) {
        float acc = b3[tid];
        #pragma unroll 8
        for (int ii = 0; ii < T_; ++ii) acc += bufA[ii] * w3[ii * I_ + tid];
        s_out[b * I_ + tid] = acc + 1.0f;
    }
}

// ---------------------------------------------------------------------------
// Kernel 2: modulate + demodulate, emit f16 weights wprep[b][k][o][i]
// one block per (o, b)
// ---------------------------------------------------------------------------
__global__ void __launch_bounds__(256)
modconv_demod_kernel(const float* __restrict__ conv_w,
                     const float* __restrict__ s,
                     _Float16* __restrict__ wprep) {
    __shared__ float red[256];
    const int o = blockIdx.x;
    const int b = blockIdx.y;
    const int i = threadIdx.x;   // 256 threads == I_

    const float sv = s[b * I_ + i];
    float m0 = conv_w[(o * I_ + i) * K_ + 0] * sv;
    float m1 = conv_w[(o * I_ + i) * K_ + 1] * sv;
    float m2 = conv_w[(o * I_ + i) * K_ + 2] * sv;

    red[i] = m0 * m0 + m1 * m1 + m2 * m2;
    __syncthreads();
    #pragma unroll
    for (int stride = 128; stride > 0; stride >>= 1) {
        if (i < stride) red[i] += red[i + stride];
        __syncthreads();
    }
    const float demod = rsqrtf(red[0] + 1e-8f);
    __syncthreads();

    const size_t base = (((size_t)b * K_) * O_ + o) * I_;   // [b][k][o][i]
    wprep[base + 0 * (size_t)O_ * I_ + i] = (_Float16)(m0 * demod);
    wprep[base + 1 * (size_t)O_ * I_ + i] = (_Float16)(m1 * demod);
    wprep[base + 2 * (size_t)O_ * I_ + i] = (_Float16)(m2 * demod);
}

// ---------------------------------------------------------------------------
// WMMA fragment loaders (wave32 layouts per CDNA5 ISA 7.12.2)
// ---------------------------------------------------------------------------
__device__ __forceinline__ v16h load_frag_a(const _Float16* ldsA, int row, int g) {
    // A 16x32 f16: lane<16 holds K 0-7 & 16-23; lane>=16 holds K 8-15 & 24-31
    const _Float16* p = ldsA + row * ASTRIDE + g * 8;
    v8h lo = *(const v8h*)p;          // halfs 0..7  -> K = g*8 + 0..7
    v8h hi = *(const v8h*)(p + 16);   // halfs 8..15 -> K = 16 + g*8 + 0..7
    return __builtin_shufflevector(lo, hi, 0,1,2,3,4,5,6,7,8,9,10,11,12,13,14,15);
}

__device__ __forceinline__ v16h load_frag_b(const _Float16* ldsB, int row, int g) {
    // B 32x16 f16: lane<16 holds K 0-7 & 16-23 (N=lane); lane>=16 holds K 8-15 & 24-31
    const _Float16* p = ldsB + row * BSTRIDE + g * 8;
    v8h lo = *(const v8h*)p;
    v8h hi = *(const v8h*)(p + 16);
    return __builtin_shufflevector(lo, hi, 0,1,2,3,4,5,6,7,8,9,10,11,12,13,14,15);
}

// ---------------------------------------------------------------------------
// Kernel 3: conv1d as 3 shifted GEMMs via v_wmma_f32_16x16x32_f16
// grid: (L/128, O/128, B); block: 256 threads = 8 waves
// block tile: 128(O) x 128(L); wave tile: 64(O) x 32(L) -> 8 f32 accumulators
// ---------------------------------------------------------------------------
__global__ void __launch_bounds__(256)
modconv_conv_kernel(const float* __restrict__ bx,
                    const _Float16* __restrict__ wprep,
                    const float* __restrict__ conv_b,
                    float* __restrict__ out) {
    __shared__ __align__(16) _Float16 ldsA[K_ * 128 * ASTRIDE]; // [k][m][i]  30720 B
    __shared__ __align__(16) _Float16 ldsB[130 * BSTRIDE];      // [l+halo][i] 10400 B

    const int l0 = blockIdx.x * 128;
    const int o0 = blockIdx.y * 128;
    const int b  = blockIdx.z;

    const int tid    = threadIdx.x;
    const int lane   = tid & 31;
    const int wave   = tid >> 5;      // 0..7
    const int wave_m = wave & 1;      // 0..1  -> m base = wave_m*64
    const int wave_n = wave >> 1;     // 0..3  -> n base = wave_n*32
    const int g      = lane >> 4;     // lane half-group
    const int ln     = lane & 15;

    v8f acc[4][2];
    #pragma unroll
    for (int mt = 0; mt < 4; ++mt)
        #pragma unroll
        for (int nt = 0; nt < 2; ++nt)
            acc[mt][nt] = (v8f){};

    for (int ic = 0; ic < I_; ic += 32) {
        __syncthreads();

        // ---- stage activations: bx[b, ic..ic+31, l0-1 .. l0+128] -> ldsB[l][i] (f16, transposed)
        for (int e = tid; e < 32 * 130; e += 256) {
            const int ki = e / 130;
            const int n  = e % 130;
            const int l  = l0 - 1 + n;
            float v = 0.0f;
            if (l >= 0 && l < L_)
                v = bx[((size_t)b * I_ + (ic + ki)) * L_ + l];
            ldsB[n * BSTRIDE + ki] = (_Float16)v;
        }

        // ---- stage weights: wprep[b][k][o0+m][ic..ic+31] -> ldsA[k][m][i], 16B chunks
        for (int e = tid; e < 1536; e += 256) {
            const int k   = e >> 9;        // 0..2
            const int rem = e & 511;
            const int m   = rem >> 2;      // 0..127
            const int c   = rem & 3;       // 16B chunk in 64B row
            const v8h* src = (const v8h*)(wprep +
                ((((size_t)b * K_ + k) * O_ + (o0 + m)) * I_ + ic) + c * 8);
            *(v8h*)&ldsA[(k * 128 + m) * ASTRIDE + c * 8] = *src;
        }

        // prefetch next activation chunk into cache (global_prefetch_b8)
        if (ic + 32 < I_)
            __builtin_prefetch(bx + ((size_t)b * I_ + (ic + 32)) * L_ + l0 + lane * 4, 0, 0);

        __syncthreads();

        // ---- compute: 3 shifts x 4 m-subtiles x 2 n-subtiles = 24 WMMAs / wave / step
        #pragma unroll
        for (int k = 0; k < K_; ++k) {
            const v16h bf0 = load_frag_b(ldsB, wave_n * 32 +  0 + ln + k, g);
            const v16h bf1 = load_frag_b(ldsB, wave_n * 32 + 16 + ln + k, g);
            #pragma unroll
            for (int mt = 0; mt < 4; ++mt) {
                const v16h af = load_frag_a(ldsA, (k * 128) + wave_m * 64 + mt * 16 + ln, g);
                acc[mt][0] = __builtin_amdgcn_wmma_f32_16x16x32_f16(
                    false, af, false, bf0, (short)0, acc[mt][0], false, false);
                acc[mt][1] = __builtin_amdgcn_wmma_f32_16x16x32_f16(
                    false, af, false, bf1, (short)0, acc[mt][1], false, false);
            }
        }
    }

    // ---- store: C/D layout -> VGPR r: M = r + 8*g, N = ln
    #pragma unroll
    for (int mt = 0; mt < 4; ++mt) {
        const int obase = o0 + wave_m * 64 + mt * 16 + g * 8;
        #pragma unroll
        for (int nt = 0; nt < 2; ++nt) {
            const int lcol = l0 + wave_n * 32 + nt * 16 + ln;
            #pragma unroll
            for (int r = 0; r < 8; ++r) {
                const int o = obase + r;
                out[((size_t)b * O_ + o) * L_ + lcol] = acc[mt][nt][r] + conv_b[o];
            }
        }
    }
}

// ---------------------------------------------------------------------------
extern "C" void kernel_launch(void* const* d_in, const int* in_sizes, int n_in,
                              void* d_out, int out_size, void* d_ws, size_t ws_size,
                              hipStream_t stream) {
    const float* bx     = (const float*)d_in[0];
    const float* t      = (const float*)d_in[1];
    const float* conv_w = (const float*)d_in[2];
    const float* conv_b = (const float*)d_in[3];
    const float* w1     = (const float*)d_in[4];
    const float* b1     = (const float*)d_in[5];
    const float* w2     = (const float*)d_in[6];
    const float* b2     = (const float*)d_in[7];
    const float* w3     = (const float*)d_in[8];
    const float* b3     = (const float*)d_in[9];
    float* out = (float*)d_out;

    float*    s_ws  = (float*)d_ws;                            // 8KB
    _Float16* wprep = (_Float16*)((char*)d_ws + 8192);         // ~3MB, [b][k][o][i]

    modconv_mlp_kernel<<<dim3(B_), dim3(256), 0, stream>>>(
        t, w1, b1, w2, b2, w3, b3, s_ws);

    modconv_demod_kernel<<<dim3(O_, B_), dim3(256), 0, stream>>>(
        conv_w, s_ws, wprep);

    modconv_conv_kernel<<<dim3(L_ / 128, O_ / 128, B_), dim3(256), 0, stream>>>(
        bx, wprep, conv_b, out);
}
